// Encoder_18348100289268
// MI455X (gfx1250) — matrix operand order repaired and checked by reading, not compiled, not run on hardware
//
#include <hip/hip_runtime.h>

typedef __attribute__((ext_vector_type(16))) _Float16 v16h;
typedef __attribute__((ext_vector_type(8)))  _Float16 v8h;
typedef __attribute__((ext_vector_type(8)))  float    v8f;

#define WMMA_F16(a, b, c) \
  __builtin_amdgcn_wmma_f32_16x16x32_f16(false, (a), false, (b), (short)0, (c), false, false)

constexpr int S_LEN   = 1024;
constexpr int D_MODEL = 768;
constexpr int NHEAD   = 12;
constexpr int HD      = 64;
constexpr int BATCH   = 8;
constexpr int M_TOK   = BATCH * S_LEN;   // 8192 tokens
constexpr float EPSV  = 1e-5f;

// -------------------------------------------------------------------------
// 16-lane (half-wave) max butterfly. WMMA C-layout keeps one matrix row in
// 16 lanes of a wave row, and v_permlane16_b32 permutes each 16-lane row
// identically -> pure-VALU reduction, no DS round-trips.
// -------------------------------------------------------------------------
__device__ __forceinline__ float rowmax16(float v)
{
#if defined(__has_builtin) && __has_builtin(__builtin_amdgcn_permlane16)
  int x;
  x = __builtin_amdgcn_permlane16(__float_as_int(v), __float_as_int(v),
                                  0x67452301u, 0xEFCDAB89u, false, false); // ^1
  v = fmaxf(v, __int_as_float(x));
  x = __builtin_amdgcn_permlane16(__float_as_int(v), __float_as_int(v),
                                  0x54761032u, 0xDCFE98BAu, false, false); // ^2
  v = fmaxf(v, __int_as_float(x));
  x = __builtin_amdgcn_permlane16(__float_as_int(v), __float_as_int(v),
                                  0x32107654u, 0xBA98FEDCu, false, false); // ^4
  v = fmaxf(v, __int_as_float(x));
  x = __builtin_amdgcn_permlane16(__float_as_int(v), __float_as_int(v),
                                  0xFEDCBA98u, 0x76543210u, false, false); // ^8
  v = fmaxf(v, __int_as_float(x));
#else
  v = fmaxf(v, __shfl_xor(v, 1));
  v = fmaxf(v, __shfl_xor(v, 2));
  v = fmaxf(v, __shfl_xor(v, 4));
  v = fmaxf(v, __shfl_xor(v, 8));
#endif
  return v;
}

// -------------------------------------------------------------------------
// h = 2*emb[x] + P   (P = sin/cos positional encoding, computed on the fly)
// -------------------------------------------------------------------------
__global__ void embed_pos_kernel(const int* __restrict__ x,
                                 const float* __restrict__ emb,
                                 float* __restrict__ h)
{
  int i = blockIdx.x * 256 + threadIdx.x;
  if (i >= M_TOK * D_MODEL) return;
  int tok = i / D_MODEL;
  int d   = i - tok * D_MODEL;
  int s   = tok & (S_LEN - 1);
  float w   = __powf(10000.0f, -(float)(d & ~1) / (float)D_MODEL);
  float ang = (float)s * w;
  float p   = (d & 1) ? __cosf(ang) : __sinf(ang);
  h[i] = 2.0f * emb[(size_t)x[tok] * D_MODEL + d] + p;
}

// -------------------------------------------------------------------------
// One-time weight convert+transpose:  W[K][768] f32  ->  Wt[768][K] f16
// -------------------------------------------------------------------------
__global__ void wt_convert_kernel(const float* __restrict__ W,
                                  _Float16* __restrict__ Wt, int K)
{
  int i = blockIdx.x * 256 + threadIdx.x;
  if (i >= K * D_MODEL) return;
  int n = i / K;
  int k = i - n * K;
  Wt[i] = (_Float16)W[(size_t)k * D_MODEL + n];
}

// -------------------------------------------------------------------------
// Tiled WMMA GEMM:  out = act(A[M,K] @ W[K,N] + bias)
//   A: f32 row-major (lda), Wt: f16 pre-transposed [N][K]
//   block = 128 threads (4 waves), tile 128(M) x 64(N), K-step 32
//   per wave: 32 rows x 64 cols -> 8 WMMA per K-step
//   MODE 0: f32 row-major   MODE 1: f16 row-major   MODE 2: f16 transposed
// -------------------------------------------------------------------------
template<int MODE, bool RELU>
__global__ void __launch_bounds__(128)
gemm_kernel(const float* __restrict__ A, int lda,
            const _Float16* __restrict__ Wt, const float* __restrict__ bias,
            int K, void* __restrict__ out)
{
  __shared__ alignas(32) _Float16 As[128 * 40];  // A tile f16, 80B row pitch
  __shared__ alignas(32) _Float16 Ws[64 * 40];   // W tile [n][k]

  const int tid  = threadIdx.x;
  const int wave = tid >> 5;
  const int lane = tid & 31;
  const int lh   = lane >> 4;     // lane half (0/1)
  const int lm   = lane & 15;
  const int bm   = blockIdx.x;
  const int bn   = blockIdx.y;

  v8f acc[2][4];
  for (int j = 0; j < 2; ++j)
    for (int t = 0; t < 4; ++t)
      for (int r = 0; r < 8; ++r) acc[j][t][r] = 0.0f;

  const int wrow = tid >> 1, whalf = tid & 1;    // W staging: 64 rows x 2 halves

  for (int k0 = 0; k0 < K; k0 += 32) {
    if (k0 + 32 < K) {   // prefetch next K-tile (global_prefetch_b8)
      __builtin_prefetch(A + (size_t)(bm * 128 + tid) * lda + k0 + 32, 0, 1);
      __builtin_prefetch(Wt + (size_t)(bn * 64 + wrow) * K + k0 + 32 + whalf * 16, 0, 1);
    }
    // stage A: each thread converts one full 32-wide row
    {
      const float* src = A + (size_t)(bm * 128 + tid) * lda + k0;
      _Float16 tmp[32];
      #pragma unroll
      for (int i = 0; i < 32; ++i) tmp[i] = (_Float16)src[i];
      v8h* dst = (v8h*)&As[tid * 40];
      #pragma unroll
      for (int i = 0; i < 4; ++i) dst[i] = ((v8h*)tmp)[i];
    }
    // stage W: pure f16 copy (pre-transposed in global)
    {
      const _Float16* src = Wt + (size_t)(bn * 64 + wrow) * K + k0 + whalf * 16;
      _Float16* dst = &Ws[wrow * 40 + whalf * 16];
      *(v8h*)(dst)     = *(const v8h*)(src);
      *(v8h*)(dst + 8) = *(const v8h*)(src + 8);
    }
    __syncthreads();

    v16h afrag[2];
    #pragma unroll
    for (int j = 0; j < 2; ++j) {
      const _Float16* base = &As[(32 * wave + 16 * j + lm) * 40];
      ((v8h*)&afrag[j])[0] = *(const v8h*)(base + 16 * lh);
      ((v8h*)&afrag[j])[1] = *(const v8h*)(base + 16 * lh + 8);
    }
    #pragma unroll
    for (int nt = 0; nt < 4; ++nt) {
      v16h bfrag;
      const _Float16* base = &Ws[(16 * nt + lm) * 40 + 16 * lh];
      ((v8h*)&bfrag)[0] = *(const v8h*)(base);
      ((v8h*)&bfrag)[1] = *(const v8h*)(base + 8);
      acc[0][nt] = WMMA_F16(afrag[0], bfrag, acc[0][nt]);
      acc[1][nt] = WMMA_F16(afrag[1], bfrag, acc[1][nt]);
    }
    __syncthreads();
  }

  #pragma unroll
  for (int j = 0; j < 2; ++j) {
    const int rowbase = bm * 128 + 32 * wave + 16 * j + 8 * lh;
    #pragma unroll
    for (int nt = 0; nt < 4; ++nt) {
      const int col = bn * 64 + 16 * nt + lm;
      const float bv = bias[col];
      if (MODE == 0) {
        float* o = (float*)out;
        #pragma unroll
        for (int r = 0; r < 8; ++r) {
          float v = acc[j][nt][r] + bv;
          if (RELU) v = v > 0.0f ? v : 0.0f;
          o[(size_t)(rowbase + r) * D_MODEL + col] = v;
        }
      } else if (MODE == 1) {
        _Float16* o = (_Float16*)out;
        #pragma unroll
        for (int r = 0; r < 8; ++r) {
          float v = acc[j][nt][r] + bv;
          if (RELU) v = v > 0.0f ? v : 0.0f;
          o[(size_t)(rowbase + r) * D_MODEL + col] = (_Float16)v;
        }
      } else {  // f16 transposed [col][row]: 8 consecutive rows -> one b128 store
        _Float16* o = (_Float16*)out;
        v8h pack;
        #pragma unroll
        for (int r = 0; r < 8; ++r) {
          float v = acc[j][nt][r] + bv;
          if (RELU) v = v > 0.0f ? v : 0.0f;
          pack[r] = (_Float16)v;
        }
        *(v8h*)(o + (size_t)col * M_TOK + rowbase) = pack;
      }
    }
  }
}

// -------------------------------------------------------------------------
// Flash-style attention. 1 wave = 16-query tile of one (batch, head).
// Q,K: f16 [token][768] row-major; V: f16 transposed [d][token].
// 32 keys/iter: 4 WMMA scores, 1 WMMA row-sum (P x ones), 4 WMMA P@V.
// 1/sqrt(768) scale folded into the Q fragment; max reduce via permlane16.
// -------------------------------------------------------------------------
__global__ void __launch_bounds__(128)
attention_kernel(const _Float16* __restrict__ Q,
                 const _Float16* __restrict__ Km,
                 const _Float16* __restrict__ Vt,
                 float* __restrict__ Ov)
{
  __shared__ alignas(32) _Float16 plds[4][16 * 40];
  const int tid  = threadIdx.x;
  const int wave = tid >> 5;
  const int lane = tid & 31;
  const int lh   = lane >> 4;
  const int lm   = lane & 15;

  const int id   = blockIdx.x * 4 + wave;   // 0 .. B*H*(S/16)-1
  const int qt   = id & 63;
  const int bh   = id >> 6;
  const int head = bh % NHEAD;
  const int b    = bh / NHEAD;

  const float scale = 0.036084391824351615f;  // 1/sqrt(768)

  // resident Q A-fragments, pre-scaled
  v16h qf[2];
  {
    const _Float16* qrow = Q + (size_t)(b * S_LEN + qt * 16 + lm) * D_MODEL + head * HD;
    #pragma unroll
    for (int c = 0; c < 2; ++c) {
      ((v8h*)&qf[c])[0] = *(const v8h*)(qrow + 32 * c + 16 * lh);
      ((v8h*)&qf[c])[1] = *(const v8h*)(qrow + 32 * c + 16 * lh + 8);
      #pragma unroll
      for (int i = 0; i < 16; ++i)
        qf[c][i] = (_Float16)((float)qf[c][i] * scale);
    }
  }

  v16h ones;
  #pragma unroll
  for (int i = 0; i < 16; ++i) ones[i] = (_Float16)1.0f;

  v8f oacc[4];
  for (int t = 0; t < 4; ++t)
    for (int r = 0; r < 8; ++r) oacc[t][r] = 0.0f;
  float mrow[8], lrow[8];
  for (int r = 0; r < 8; ++r) { mrow[r] = -1e30f; lrow[r] = 0.0f; }

  const _Float16* kbase = Km + (size_t)(b * S_LEN) * D_MODEL + head * HD;
  const _Float16* vbase = Vt + (size_t)(head * HD) * M_TOK + b * S_LEN;
  _Float16* pw = plds[wave];

  for (int kt = 0; kt < S_LEN / 32; ++kt) {
    // two 16x16 score tiles (keys kt*32 .. kt*32+31)
    v8f sc[2];
    #pragma unroll
    for (int st = 0; st < 2; ++st) {
      for (int r = 0; r < 8; ++r) sc[st][r] = 0.0f;
      const _Float16* krow = kbase + (size_t)(kt * 32 + st * 16 + lm) * D_MODEL;
      #pragma unroll
      for (int c = 0; c < 2; ++c) {
        v16h kf = *(const v16h*)(krow + 32 * c + 16 * lh);
        sc[st] = WMMA_F16(qf[c], kf, sc[st]);
      }
    }

    // online softmax: max via permlane16 butterflies, sum via WMMA below
    float corr[8];
    #pragma unroll
    for (int r = 0; r < 8; ++r) {
      float rm = rowmax16(fmaxf(sc[0][r], sc[1][r]));
      float mnew = fmaxf(mrow[r], rm);
      corr[r] = __expf(mrow[r] - mnew);
      float p0 = __expf(sc[0][r] - mnew);
      float p1 = __expf(sc[1][r] - mnew);
      mrow[r] = mnew;
      int row = r + 8 * lh;
      pw[row * 40 + lm]      = (_Float16)p0;
      pw[row * 40 + 16 + lm] = (_Float16)p1;
      #pragma unroll
      for (int t = 0; t < 4; ++t) oacc[t][r] *= corr[r];
    }
    __syncthreads();

    // re-layout P (C-layout -> A-fragment) via LDS
    v16h pf;
    {
      const _Float16* base = &pw[lm * 40];
      ((v8h*)&pf)[0] = *(const v8h*)(base + 16 * lh);
      ((v8h*)&pf)[1] = *(const v8h*)(base + 16 * lh + 8);
    }
    // row-sum of P via WMMA with ones: each lane gets its rows' sums
    {
      v8f rs;
      for (int r = 0; r < 8; ++r) rs[r] = 0.0f;
      rs = WMMA_F16(pf, ones, rs);
      #pragma unroll
      for (int r = 0; r < 8; ++r) lrow[r] = lrow[r] * corr[r] + rs[r];
    }
    // P @ V
    #pragma unroll
    for (int nt = 0; nt < 4; ++nt) {
      v16h vf = *(const v16h*)(vbase + (size_t)(16 * nt + lm) * M_TOK + kt * 32 + 16 * lh);
      oacc[nt] = WMMA_F16(pf, vf, oacc[nt]);
    }
    __syncthreads();
  }

  float linv[8];
  #pragma unroll
  for (int r = 0; r < 8; ++r) linv[r] = 1.0f / lrow[r];
  #pragma unroll
  for (int nt = 0; nt < 4; ++nt) {
    int col = head * HD + 16 * nt + lm;
    #pragma unroll
    for (int r = 0; r < 8; ++r) {
      int row = b * S_LEN + qt * 16 + r + 8 * lh;
      Ov[(size_t)row * D_MODEL + col] = oacc[nt][r] * linv[r];
    }
  }
}

// -------------------------------------------------------------------------
// BatchNorm over (batch*seq) per feature, training mode, biased variance.
// -------------------------------------------------------------------------
__global__ void zero_kernel(float* p, int n)
{
  int i = blockIdx.x * 256 + threadIdx.x;
  if (i < n) p[i] = 0.0f;
}

__global__ void __launch_bounds__(256)
bn_stats_kernel(const float* __restrict__ a, const float* __restrict__ b,
                float* __restrict__ sums, float* __restrict__ sq)
{
  const int r0 = blockIdx.x * 512;   // 16 blocks x 512 rows
  for (int c = threadIdx.x; c < D_MODEL; c += 256) {
    float s = 0.0f, q = 0.0f;
    for (int r = r0; r < r0 + 512; ++r) {
      float v = a[(size_t)r * D_MODEL + c] + b[(size_t)r * D_MODEL + c];
      s += v;
      q += v * v;
    }
    atomicAdd(&sums[c], s);
    atomicAdd(&sq[c], q);
  }
}

__global__ void bn_apply_kernel(const float* __restrict__ a, const float* __restrict__ b,
                                const float* __restrict__ sums, const float* __restrict__ sq,
                                const float* __restrict__ gamma, const float* __restrict__ beta,
                                float* __restrict__ out)
{
  int i = blockIdx.x * 256 + threadIdx.x;
  if (i >= M_TOK * D_MODEL) return;
  int c = i % D_MODEL;
  const float invN = 1.0f / (float)M_TOK;
  float mean = sums[c] * invN;
  float var  = sq[c] * invN - mean * mean;
  float v = a[i] + b[i];
  out[i] = (v - mean) * rsqrtf(var + EPSV) * gamma[c] + beta[c];
}

// -------------------------------------------------------------------------
extern "C" void kernel_launch(void* const* d_in, const int* in_sizes, int n_in,
                              void* d_out, int out_size, void* d_ws, size_t ws_size,
                              hipStream_t stream)
{
  const int*   x   = (const int*)  d_in[0];
  const float* emb = (const float*)d_in[1];
  const float* Wq  = (const float*)d_in[2];
  const float* bq  = (const float*)d_in[3];
  const float* Wk  = (const float*)d_in[4];
  const float* bk  = (const float*)d_in[5];
  const float* Wv  = (const float*)d_in[6];
  const float* bv  = (const float*)d_in[7];
  const float* Wf  = (const float*)d_in[8];
  const float* bf  = (const float*)d_in[9];
  const float* g1  = (const float*)d_in[10];
  const float* b1  = (const float*)d_in[11];
  const float* g2  = (const float*)d_in[12];
  const float* b2  = (const float*)d_in[13];

  char* ws = (char*)d_ws;
  size_t off = 0;
  const size_t f32mat = (size_t)M_TOK * D_MODEL * sizeof(float);
  const size_t f16mat = (size_t)M_TOK * D_MODEL * sizeof(_Float16);
  float*     h    = (float*)(ws + off);     off += f32mat;
  float*     t    = (float*)(ws + off);     off += f32mat;
  float*     y    = (float*)(ws + off);     off += f32mat;   // attn out, then FF out
  _Float16*  Qh   = (_Float16*)(ws + off);  off += f16mat;
  _Float16*  Kh   = (_Float16*)(ws + off);  off += f16mat;
  _Float16*  Vt   = (_Float16*)(ws + off);  off += f16mat;   // transposed [d][token]
  _Float16*  Wqt  = (_Float16*)(ws + off);  off += (size_t)256 * D_MODEL * 2;
  _Float16*  Wkt  = (_Float16*)(ws + off);  off += (size_t)256 * D_MODEL * 2;
  _Float16*  Wvt  = (_Float16*)(ws + off);  off += (size_t)256 * D_MODEL * 2;
  _Float16*  Wft  = (_Float16*)(ws + off);  off += (size_t)D_MODEL * D_MODEL * 2;
  float*     sums = (float*)(ws + off);     off += D_MODEL * sizeof(float);
  float*     sq   = (float*)(ws + off);     off += D_MODEL * sizeof(float);

  const int nel = M_TOK * D_MODEL;
  dim3 gemmGrid(M_TOK / 128, D_MODEL / 64);
  const int attnBlocks = BATCH * NHEAD * (S_LEN / 16) / 4;

  // one-time weight convert+transpose (shared by both layers)
  wt_convert_kernel<<<(256 * D_MODEL + 255) / 256, 256, 0, stream>>>(Wq, Wqt, 256);
  wt_convert_kernel<<<(256 * D_MODEL + 255) / 256, 256, 0, stream>>>(Wk, Wkt, 256);
  wt_convert_kernel<<<(256 * D_MODEL + 255) / 256, 256, 0, stream>>>(Wv, Wvt, 256);
  wt_convert_kernel<<<(D_MODEL * D_MODEL + 255) / 256, 256, 0, stream>>>(Wf, Wft, D_MODEL);

  embed_pos_kernel<<<(nel + 255) / 256, 256, 0, stream>>>(x, emb, h);

  for (int layer = 0; layer < 2; ++layer) {
    // Q/K/V projections + ReLU (K row-major f16, V transposed f16)
    gemm_kernel<1, true ><<<gemmGrid, 128, 0, stream>>>(h + 0,   D_MODEL, Wqt, bq, 256, Qh);
    gemm_kernel<1, true ><<<gemmGrid, 128, 0, stream>>>(h + 256, D_MODEL, Wkt, bk, 256, Kh);
    gemm_kernel<2, true ><<<gemmGrid, 128, 0, stream>>>(h + 512, D_MODEL, Wvt, bv, 256, Vt);

    attention_kernel<<<attnBlocks, 128, 0, stream>>>(Qh, Kh, Vt, y);

    // t = BN(h + attn_out)
    zero_kernel<<<6, 256, 0, stream>>>(sums, 2 * D_MODEL);  // sums & sq contiguous
    bn_stats_kernel<<<16, 256, 0, stream>>>(h, y, sums, sq);
    bn_apply_kernel<<<(nel + 255) / 256, 256, 0, stream>>>(h, y, sums, sq, g1, b1, t);

    // y = t @ Wf + bf
    gemm_kernel<0, false><<<gemmGrid, 128, 0, stream>>>(t, D_MODEL, Wft, bf, D_MODEL, y);

    // h = BN(t + y)   (final layer writes to d_out)
    zero_kernel<<<6, 256, 0, stream>>>(sums, 2 * D_MODEL);
    bn_stats_kernel<<<16, 256, 0, stream>>>(t, y, sums, sq);
    float* dst = (layer == 1) ? (float*)d_out : h;
    bn_apply_kernel<<<(nel + 255) / 256, 256, 0, stream>>>(t, y, sums, sq, g2, b2, dst);
  }
}
